// LearnablePixelwiseAnisoJBU_NoParent_28535762715334
// MI455X (gfx1250) — compile-verified, matrix operand order checked by reading, not run on hardware
//
#include <hip/hip_runtime.h>
#include <hip/hip_bf16.h>
#include <math.h>

// ---- CDNA5 WMMA vector types ----
typedef _Float16 v16h __attribute__((ext_vector_type(16)));
typedef _Float16 v8h  __attribute__((ext_vector_type(8)));
typedef float    v8f  __attribute__((ext_vector_type(8)));

#define NB    2      // batch
#define NC    64     // channels
#define HL    16
#define WL    16
#define HH    128
#define WH    128
#define SCALE 8
#define RMAXI 4
#define KTAP  81     // 9x9 taps
#define KPAD  96     // padded K for 3 x (16x16x32) WMMA steps
#define FSTR  104    // half stride for s_feat / s_w rows (bank-conflict pad)
#define LSTR  88     // float stride for s_logw rows

// Load a 16-half A/B fragment (row-major-in-K LDS row) per the ISA 16-bit
// 16x32 layout: lane-hi=0 -> K = ks*32 + {0..7, 16..23}; lane-hi=1 -> +8.
__device__ __forceinline__ v16h ld_frag(const _Float16* __restrict__ row, int hi, int ks) {
    const _Float16* p0 = row + ks * 32 + hi * 8;
    v8h a0 = *(const v8h*)(p0);
    v8h a1 = *(const v8h*)(p0 + 16);
    return __builtin_shufflevector(a0, a1, 0,1,2,3,4,5,6,7,8,9,10,11,12,13,14,15);
}

__global__ __launch_bounds__(256)
void aniso_jbu_wmma_kernel(const float* __restrict__ feat,
                           const float* __restrict__ guide,
                           const float* __restrict__ sxr,
                           const float* __restrict__ syr,
                           const float* __restrict__ thr,
                           const float* __restrict__ srr,
                           float* __restrict__ out)
{
    __shared__ __align__(16) _Float16 s_feat[NC][FSTR]; // A: feat_g[c][k] (f16)
    __shared__ __align__(16) _Float16 s_w[64][FSTR];    // B: s[p][k]      (f16)
    __shared__ float s_logw[64][LSTR];                  // masked log-weights
    __shared__ float t_cos[KPAD], t_sin[KPAD];
    __shared__ float t_i2x[KPAD], t_i2y[KPAD], t_i2r[KPAD];
    __shared__ float t_cx[KPAD], t_cy[KPAD];
    __shared__ float t_g0[KPAD], t_g1[KPAD], t_g2[KPAD];
    __shared__ int   t_rad2[KPAD];
    __shared__ float p_g[3][64];
    __shared__ int   p_R2[64];
    __shared__ float p_den[64];
    __shared__ float red_max[64][4];
    __shared__ float red_den[64][4];

    const int t  = threadIdx.x;
    const int b  = blockIdx.z;
    const int ty = blockIdx.y;   // LR row index == uc for the whole 8x8 tile
    const int tx = blockIdx.x;   // LR col index == vc

    // ---------------- phase 0a: per-tap shared parameters ----------------
    if (t < KTAP) {
        const int k  = t;
        const int dY = k / 9 - RMAXI;
        const int dX = k % 9 - RMAXI;
        const int Ui = min(max(ty + dY, 0), HL - 1);
        const int Vi = min(max(tx + dX, 0), WL - 1);
        const int li = (b * HL + Ui) * WL + Vi;          // (B,1,HL,WL) maps
        const float sx = fmaxf(__expf(sxr[li]), 1e-6f);
        const float sy = fmaxf(__expf(syr[li]), 1e-6f);
        const float th = 3.14159265358979323846f * tanhf(thr[li]);
        const float sr = fmaxf(__expf(srr[li]), 1e-6f);
        t_cos[k]  = __cosf(th);
        t_sin[k]  = __sinf(th);
        t_i2x[k]  = 1.0f / (2.0f * sx * sx + 1e-8f);
        t_i2y[k]  = 1.0f / (2.0f * sy * sy + 1e-8f);
        t_i2r[k]  = 1.0f / (2.0f * sr * sr + 1e-8f);
        const float cy = (Ui + 0.5f) * SCALE - 0.5f;
        const float cx = (Vi + 0.5f) * SCALE - 0.5f;
        t_cy[k] = cy;  t_cx[k] = cx;
        t_rad2[k] = dY * dY + dX * dX;
        // guide_lr[(c),Ui,Vi]: bilinear downsample of guide_hr (same coords as cy/cx)
        int y0 = min(max((int)floorf(cy), 0), HH - 1);
        int y1 = min(y0 + 1, HH - 1);
        int x0 = min(max((int)floorf(cx), 0), WH - 1);
        int x1 = min(x0 + 1, WH - 1);
        const float wy = cy - (float)y0;
        const float wx = cx - (float)x0;
        #pragma unroll
        for (int c = 0; c < 3; ++c) {
            const float* gc = guide + ((size_t)(b * 3 + c)) * HH * WH;
            float g00 = gc[y0 * WH + x0], g01 = gc[y0 * WH + x1];
            float g10 = gc[y1 * WH + x0], g11 = gc[y1 * WH + x1];
            float g = (g00 * (1.f - wy) + g10 * wy) * (1.f - wx)
                    + (g01 * (1.f - wy) + g11 * wy) * wx;
            if (c == 0) t_g0[k] = g; else if (c == 1) t_g1[k] = g; else t_g2[k] = g;
        }
    }

    // ---------------- phase 0b: per-pixel shared parameters ----------------
    if (t < 64) {
        const int p = t;
        const int Y = ty * 8 + (p >> 3);
        const int X = tx * 8 + (p & 7);
        #pragma unroll
        for (int c = 0; c < 3; ++c)
            p_g[c][p] = guide[((size_t)(b * 3 + c) * HH + Y) * WH + X];
        // sigma_eff_hr: bilinear resize of max(exp(sx),exp(sy)) to HR
        float u = fminf(fmaxf((Y + 0.5f) * (1.0f / SCALE) - 0.5f, 0.f), (float)(HL - 1));
        float v = fminf(fmaxf((X + 0.5f) * (1.0f / SCALE) - 0.5f, 0.f), (float)(WL - 1));
        int y0 = (int)floorf(u), x0 = (int)floorf(v);
        int y1 = min(y0 + 1, HL - 1), x1 = min(x0 + 1, WL - 1);
        float wy = u - (float)y0, wx = v - (float)x0;
        auto se = [&](int i, int j) {
            int id = (b * HL + i) * WL + j;
            return fmaxf(__expf(sxr[id]), __expf(syr[id]));
        };
        float s00 = se(y0, x0), s01 = se(y0, x1), s10 = se(y1, x0), s11 = se(y1, x1);
        float seff = (s00 * (1.f - wy) + s10 * wy) * (1.f - wx)
                   + (s01 * (1.f - wy) + s11 * wy) * wx;
        int R = (int)ceilf(2.0f * seff);
        R = min(max(R, 1), RMAXI);
        p_R2[p] = R * R;
    }

    // ---------------- phase 0c: stage feat taps into LDS (f16 A matrix) ----------------
    {
        const int c    = t & 63;
        const int slot = t >> 6;
        for (int k = slot; k < KTAP; k += 4) {
            int dY = k / 9 - RMAXI, dX = k % 9 - RMAXI;
            int Ui = min(max(ty + dY, 0), HL - 1);
            int Vi = min(max(tx + dX, 0), WL - 1);
            s_feat[c][k] = (_Float16)feat[((size_t)(b * NC + c) * HL + Ui) * WL + Vi];
        }
        for (int k = KTAP + slot; k < KPAD; k += 4)
            s_feat[c][k] = (_Float16)0.f;   // zero K-pad -> exact 0 contribution
    }
    __syncthreads();

    // ---------------- phase 1: masked log-weights + per-pixel max ----------------
    const int p    = t & 63;
    const int slot = t >> 6;
    const float Yf = (float)(ty * 8 + (p >> 3));
    const float Xf = (float)(tx * 8 + (p & 7));
    const float g0 = p_g[0][p], g1 = p_g[1][p], g2 = p_g[2][p];
    const int   R2 = p_R2[p];
    float lmax = -3.0e38f;
    for (int k = slot; k < KTAP; k += 4) {
        float dx = Xf - t_cx[k];
        float dy = Yf - t_cy[k];
        float ct = t_cos[k], st = t_sin[k];
        float xp =  dx * ct + dy * st;
        float yp = -dx * st + dy * ct;
        float lw = -(xp * xp) * t_i2x[k] - (yp * yp) * t_i2y[k];
        float d0 = g0 - t_g0[k], d1 = g1 - t_g1[k], d2 = g2 - t_g2[k];
        lw -= (d0 * d0 + d1 * d1 + d2 * d2) * t_i2r[k];
        bool ok = (t_rad2[k] <= R2);
        lw = ok ? lw : -3.0e38f;             // masked -> exp underflows to 0
        if (ok) lmax = fmaxf(lmax, lw);
        s_logw[p][k] = lw;
    }
    red_max[p][slot] = lmax;
    __syncthreads();

    // ---------------- phase 2: s = exp(lw - m) (f16 B matrix) + den ----------------
    const float m = fmaxf(fmaxf(red_max[p][0], red_max[p][1]),
                          fmaxf(red_max[p][2], red_max[p][3]));
    float dsum = 0.f;
    for (int k = slot; k < KTAP; k += 4) {
        float s = __expf(s_logw[p][k] - m);
        dsum += s;
        s_w[p][k] = (_Float16)s;
    }
    for (int k = KTAP + slot; k < KPAD; k += 4)
        s_w[p][k] = (_Float16)0.f;
    red_den[p][slot] = dsum;
    __syncthreads();
    if (t < 64)
        p_den[t] = red_den[t][0] + red_den[t][1] + red_den[t][2] + red_den[t][3];
    __syncthreads();

    // ---------------- phase 3: GEMM num[64c x 64p] = A[64x96] x B[96x64] ----------------
    // 8 waves: wave -> M-tile mi = wave&3, two N-tiles ni0, ni0+1. EXEC all-1s here.
    const int lane = t & 31;
    const int wave = t >> 5;
    const int mi   = wave & 3;
    const int ni0  = (wave >> 2) * 2;
    const int hi   = lane >> 4;
    const int ln   = lane & 15;
    const _Float16* arow  = &s_feat[mi * 16 + ln][0];
    const _Float16* brow0 = &s_w[ni0 * 16 + ln][0];
    const _Float16* brow1 = &s_w[(ni0 + 1) * 16 + ln][0];
    v8f acc0 = {0.f, 0.f, 0.f, 0.f, 0.f, 0.f, 0.f, 0.f};
    v8f acc1 = {0.f, 0.f, 0.f, 0.f, 0.f, 0.f, 0.f, 0.f};
    #pragma unroll
    for (int ks = 0; ks < 3; ++ks) {
        v16h a  = ld_frag(arow,  hi, ks);
        v16h b0 = ld_frag(brow0, hi, ks);
        v16h b1 = ld_frag(brow1, hi, ks);
        acc0 = __builtin_amdgcn_wmma_f32_16x16x32_f16(false, a, false, b0,
                                                      (short)0, acc0, false, false);
        acc1 = __builtin_amdgcn_wmma_f32_16x16x32_f16(false, a, false, b1,
                                                      (short)0, acc1, false, false);
    }

    // ---------------- phase 4: epilogue (C/D layout: VGPR r -> M=r / 8+r) ----------------
    #pragma unroll
    for (int side = 0; side < 2; ++side) {
        const v8f acc = side ? acc1 : acc0;
        const int ni  = ni0 + side;
        const int p2  = ni * 16 + ln;
        const float den = p_den[p2];
        const int Y = ty * 8 + (p2 >> 3);
        const int X = tx * 8 + (p2 & 7);
        const float inv = 1.0f / fmaxf(den, 1e-8f);
        const bool fb = (den < 1e-6f);
        #pragma unroll
        for (int r = 0; r < 8; ++r) {
            const int c = mi * 16 + hi * 8 + r;
            float val;
            if (fb) {
                // bilinear upsample fallback of feat_lr
                float u = fminf(fmaxf((Y + 0.5f) * (1.0f / SCALE) - 0.5f, 0.f), (float)(HL - 1));
                float v = fminf(fmaxf((X + 0.5f) * (1.0f / SCALE) - 0.5f, 0.f), (float)(WL - 1));
                int y0 = (int)floorf(u), x0 = (int)floorf(v);
                int y1 = min(y0 + 1, HL - 1), x1 = min(x0 + 1, WL - 1);
                float wy = u - (float)y0, wx = v - (float)x0;
                const float* fc = feat + ((size_t)(b * NC + c)) * HL * WL;
                float f00 = fc[y0 * WL + x0], f01 = fc[y0 * WL + x1];
                float f10 = fc[y1 * WL + x0], f11 = fc[y1 * WL + x1];
                val = (f00 * (1.f - wy) + f10 * wy) * (1.f - wx)
                    + (f01 * (1.f - wy) + f11 * wy) * wx;
            } else {
                val = acc[r] * inv;
            }
            out[(((size_t)b * NC + c) * HH + Y) * WH + X] = val;
        }
    }
}

extern "C" void kernel_launch(void* const* d_in, const int* in_sizes, int n_in,
                              void* d_out, int out_size, void* d_ws, size_t ws_size,
                              hipStream_t stream) {
    (void)in_sizes; (void)n_in; (void)out_size; (void)d_ws; (void)ws_size;
    const float* feat  = (const float*)d_in[0];   // (2,64,16,16)
    const float* guide = (const float*)d_in[1];   // (2,3,128,128)
    const float* sxr   = (const float*)d_in[2];   // (2,1,16,16)
    const float* syr   = (const float*)d_in[3];
    const float* thr   = (const float*)d_in[4];
    const float* srr   = (const float*)d_in[5];
    float* out = (float*)d_out;                   // (2,64,128,128)

    dim3 grid(WH / 8, HH / 8, NB);                // one block per 8x8 HR tile
    aniso_jbu_wmma_kernel<<<grid, 256, 0, stream>>>(feat, guide, sxr, syr, thr, srr, out);
}
